// FNO1d_19696720019553
// MI455X (gfx1250) — matrix operand order repaired and checked by reading, not compiled, not run on hardware
//
#include <hip/hip_runtime.h>
#include <cstdint>
#include <cstddef>

// Problem constants (match reference)
#define B_    32
#define N_    8192
#define P_    8
#define W_    64
#define M_    32
#define L_    4
#define ROWS_ (B_ * W_)          // 2048 (b,channel) rows
#define KPART_ 8                 // K-split of the forward DFT
#define EPS_  1e-5f

typedef _Float16 v16h __attribute__((ext_vector_type(16)));
typedef _Float16 v8h  __attribute__((ext_vector_type(8)));
typedef _Float16 v4h  __attribute__((ext_vector_type(4)));
typedef float    v8f  __attribute__((ext_vector_type(8)));
typedef unsigned int v4u __attribute__((ext_vector_type(4)));
typedef int      v8i  __attribute__((ext_vector_type(8)));
typedef int      v4i  __attribute__((ext_vector_type(4)));

static __device__ __forceinline__ v16h pack16(v8h a, v8h b) {
  v16h r;
#pragma unroll
  for (int i = 0; i < 8; ++i) { r[i] = a[i]; r[8 + i] = b[i]; }
  return r;
}

static __device__ __forceinline__ v8f wmma_f16(v16h a, v16h b, v8f c) {
  // D(f32 16x16) = A(f16 16x32) * B(f16 32x16) + C
  return __builtin_amdgcn_wmma_f32_16x16x32_f16(false, a, false, b, (short)0, c, false, false);
}

static __device__ __forceinline__ float gelu_exact(float x) {
  return 0.5f * x * (1.0f + erff(x * 0.70710678118654752f));
}

// ---------------------------------------------------------------------------
// Tensor Data Mover: async copy of a 2-D f16 tile (tile_dim0=32 contiguous
// elements x tile_dim1=64 rows, row stride 8192 elements) from global into
// LDS at byte offset lds_off. Descriptor per CDNA5 ISA sec. 8.3-8.4:
//   group0: count=1 | lds_addr | global_addr(57b) | type=2
//   group1: data_size=2B, tensor_dim0=8192, tensor_dim1=64, tile=32x64,
//           tensor_dim0_stride=8192; no pad/iterate/multicast/barrier
//   group2/3: zero (2-D tensor)
// Completion tracked with TENSORcnt (s_wait_tensorcnt).
// ---------------------------------------------------------------------------
static __device__ __forceinline__ void tdm_load_b_panel(unsigned int lds_off,
                                                        const _Float16* gsrc) {
  unsigned long long ga = (unsigned long long)(uintptr_t)gsrc;
  v4u g0;
  g0[0] = 1u;                                        // count=1, user descriptor
  g0[1] = lds_off;                                   // LDS byte address
  g0[2] = (unsigned int)(ga & 0xffffffffu);          // global_addr[31:0]
  g0[3] = (unsigned int)((ga >> 32) & 0x01ffffffu)   // global_addr[56:32]
          | (2u << 30);                              // type=2 ("image")
  v8i g1;
  g1[0] = 0x10000;                 // workgroup_mask=0, data_size=1 (2 bytes)
  g1[1] = (int)(8192u << 16);      // tensor_dim0[15:0]=8192 in bits[63:48]
  g1[2] = (int)(64u << 16);        // tensor_dim0[31:16]=0 | tensor_dim1=64
  g1[3] = (int)(32u << 16);        // tensor_dim1[31:16]=0 | tile_dim0=32
  g1[4] = 64;                      // tile_dim1=64, tile_dim2=0
  g1[5] = 8192;                    // tensor_dim0_stride[31:0]
  g1[6] = 0;                       // stride hi | tensor_dim1_stride lo
  g1[7] = 0;
  v4i gz; gz[0] = 0; gz[1] = 0; gz[2] = 0; gz[3] = 0;
#if defined(__clang_major__) && (__clang_major__ >= 23)
  v8i gz8; 
#pragma unroll
  for (int i = 0; i < 8; ++i) gz8[i] = 0;
  __builtin_amdgcn_tensor_load_to_lds(g0, g1, gz, gz, gz8, 0);
#else
  __builtin_amdgcn_tensor_load_to_lds(g0, g1, gz, gz, 0);
#endif
}

// ---------------------------------------------------------------------------
// Basis tables. basT[k][n] (k-major, forward DFT B-operand / TDM source)
//               basInv[n][k] (n-major, inverse DFT B-operand, contiguous k)
// k<32: cos(2*pi*k*n/N); k>=32: sin(2*pi*(k-32)*n/N). basInv carries the 1/N
// irfft scale. Phase via exact integer mod (m*n mod N) -> small sinf/cosf args.
// ---------------------------------------------------------------------------
__global__ void init_basis_kernel(_Float16* __restrict__ basT,
                                  _Float16* __restrict__ basInv) {
  int idx = blockIdx.x * blockDim.x + threadIdx.x;   // < N_*64
  int k = idx & 63;
  int n = idx >> 6;
  int m = k & 31;
  int ph = (m * n) & (N_ - 1);
  float theta = (float)ph * (6.283185307179586f / (float)N_);
  float tv = (k < 32) ? cosf(theta) : sinf(theta);
  basT[(size_t)k * N_ + n]  = (_Float16)tv;
  basInv[(size_t)n * 64 + k] = (_Float16)(tv * (1.0f / (float)N_));
}

// W1T[h][w] = proj_w1[w][h]  (f16, B-operand layout for proj GEMM)
__global__ void prep_w1t_kernel(const float* __restrict__ w1,
                                _Float16* __restrict__ W1T) {
  int idx = blockIdx.x * blockDim.x + threadIdx.x;   // < 128*64
  int col = idx >> 6;
  int w   = idx & 63;
  W1T[(size_t)col * 64 + w] = (_Float16)w1[(size_t)w * 128 + col];
}

// ---------------------------------------------------------------------------
// Lift: per point, v[b][w][n] = Linear([coord, params(8)]) -> f16 channel-major
// ---------------------------------------------------------------------------
__global__ void lift_kernel(const float* __restrict__ params,
                            const float* __restrict__ coords,
                            const float* __restrict__ lw,
                            const float* __restrict__ lb,
                            _Float16* __restrict__ vH) {
  int idx = blockIdx.x * blockDim.x + threadIdx.x;   // < B_*N_
  int b = idx >> 13;
  int n = idx & (N_ - 1);
  float c0 = coords[idx];
  float pr[P_];
#pragma unroll
  for (int p = 0; p < P_; ++p) pr[p] = params[b * P_ + p];
#pragma unroll 4
  for (int w = 0; w < W_; ++w) {
    float a = lb[w] + c0 * lw[w];
#pragma unroll
    for (int p = 0; p < P_; ++p) a += pr[p] * lw[(p + 1) * W_ + w];
    vH[(size_t)(b * W_ + w) * N_ + n] = (_Float16)a;
  }
}

// ---------------------------------------------------------------------------
// Forward truncated DFT as WMMA GEMM: ftPart[kp] = v(2048 x 1024) * basis
//   cols 0..31 = sum v*cos ; cols 32..63 = sum v*sin   (per K partition)
// B panels (64 modes x 32 n, 4KB) stream through LDS via the Tensor Data
// Mover, double-buffered on TENSORcnt; A fragments pipelined one K-step ahead.
// Grid: (ROWS_/16 row tiles, KPART_); one wave per block. Plain stores.
// ---------------------------------------------------------------------------
__global__ void dft_fwd_kernel(const _Float16* __restrict__ vH,
                               const _Float16* __restrict__ basT,
                               float* __restrict__ ftPart) {
  __shared__ __align__(16) _Float16 bPanel[2][64 * 32];   // 2 x 4KB
  const int rowTile = blockIdx.x;         // 0..127
  const int kpart   = blockIdx.y;         // 0..KPART_-1
  const int l  = threadIdx.x;             // wave32
  const int lm = l & 15;
  const int kg = l >> 4;
  const int r0 = rowTile * 16;
  const size_t vrow = (size_t)(r0 + lm) * N_;

  const int kIters = (N_ / 32) / KPART_;  // 32
  const int n0base = kpart * kIters * 32;
  const unsigned int ldsbase = (unsigned int)(uintptr_t)&bPanel[0][0];

  // Prologue: issue TDM for panel 0, load A fragment 0
  tdm_load_b_panel(ldsbase, basT + n0base);
  v8h a0 = *reinterpret_cast<const v8h*>(vH + vrow + n0base + kg * 8);
  v8h a1 = *reinterpret_cast<const v8h*>(vH + vrow + n0base + 16 + kg * 8);
  v16h A = pack16(a0, a1);

  v8f acc[4];
#pragma unroll
  for (int t = 0; t < 4; ++t) acc[t] = (v8f)0.0f;

  for (int ki = 0; ki < kIters; ++ki) {
    const int n0 = n0base + ki * 32;
    v16h Anext = A;
    if (ki + 1 < kIters) {
      // Issue next panel + next A fragment before consuming current ones.
      tdm_load_b_panel(ldsbase + (unsigned)(((ki + 1) & 1) * (64 * 32 * 2)),
                       basT + n0 + 32);
      v8h na0 = *reinterpret_cast<const v8h*>(vH + vrow + n0 + 32 + kg * 8);
      v8h na1 = *reinterpret_cast<const v8h*>(vH + vrow + n0 + 48 + kg * 8);
      Anext = pack16(na0, na1);
      __builtin_amdgcn_s_wait_tensorcnt(1);   // current panel landed
    } else {
      __builtin_amdgcn_s_wait_tensorcnt(0);
    }
    asm volatile("" ::: "memory");            // keep LDS reads below the wait
    const _Float16* pb = &bPanel[ki & 1][0];  // [mode][32] row-major
#pragma unroll
    for (int t = 0; t < 4; ++t) {
      const int col = t * 16 + lm;
      v8h b0 = *reinterpret_cast<const v8h*>(pb + col * 32 + kg * 8);
      v8h b1 = *reinterpret_cast<const v8h*>(pb + col * 32 + 16 + kg * 8);
      acc[t] = wmma_f16(A, pack16(b0, b1), acc[t]);
    }
    A = Anext;
  }

  float* dst = ftPart + (size_t)kpart * ROWS_ * 64;
#pragma unroll
  for (int t = 0; t < 4; ++t) {
    const int col = t * 16 + lm;
#pragma unroll
    for (int r = 0; r < 8; ++r) {
      dst[(size_t)(r0 + r + kg * 8) * 64 + col] = acc[t][r];
    }
  }
}

// Sum the KPART_ partial ft buffers into ft (2048 x 64 f32)
__global__ void reduce_ft_kernel(const float* __restrict__ ftPart,
                                 float* __restrict__ ft) {
  int idx = blockIdx.x * blockDim.x + threadIdx.x;   // < ROWS_*64
  float s = 0.0f;
#pragma unroll
  for (int kp = 0; kp < KPART_; ++kp) s += ftPart[(size_t)kp * ROWS_ * 64 + idx];
  ft[idx] = s;
}

// ---------------------------------------------------------------------------
// Spectral mix (tiny VALU) + pack fused-GEMM A operand:
//   Acomb[b*64+o][0..31]  =  c_m * outR   (c_0=1, c_m=2; hermitian fold)
//   Acomb[..][32..63]     = -c_m * outI
//   Acomb[..][64..127]    = bypass_w[l][o][:]
// Also zeroes the per-row instance-norm stats for this layer.
// ---------------------------------------------------------------------------
__global__ void mix_kernel(const float* __restrict__ ft,
                           const float* __restrict__ wr,
                           const float* __restrict__ wi,
                           const float* __restrict__ bw,
                           _Float16* __restrict__ Acomb,
                           float* __restrict__ stats, int layer) {
  int idx = blockIdx.x * blockDim.x + threadIdx.x;   // < B_*W_*M_
  int m = idx & 31;
  int o = (idx >> 5) & 63;
  int b = idx >> 11;
  float oR = 0.0f, oI = 0.0f;
#pragma unroll 4
  for (int i = 0; i < W_; ++i) {
    float fR = ft[(size_t)(b * W_ + i) * 64 + m];          // sum v*cos
    float fI = ft[(size_t)(b * W_ + i) * 64 + 32 + m];     // sum v*sin
    size_t wofs = ((size_t)(layer * W_ + i) * W_ + o) * M_ + m;
    float wrv = wr[wofs], wiv = wi[wofs];
    // X = fR - i*fI ;  X * (wr + i*wi)
    oR += fR * wrv + fI * wiv;
    oI += fR * wiv - fI * wrv;
  }
  float c = (m == 0) ? 1.0f : 2.0f;
  size_t rowA = (size_t)(b * W_ + o) * 128;
  Acomb[rowA + m]      = (_Float16)(c * oR);
  Acomb[rowA + 32 + m] = (_Float16)(-c * oI);
  size_t bwo = (size_t)(layer * W_ + o) * W_;
  Acomb[rowA + 64 + m] = (_Float16)bw[bwo + m];
  Acomb[rowA + 96 + m] = (_Float16)bw[bwo + 32 + m];
  if (m == 0) {
    int r = b * W_ + o;
    stats[r] = 0.0f;
    stats[ROWS_ + r] = 0.0f;
  }
}

// ---------------------------------------------------------------------------
// Fused inverse-DFT + bypass GEMM (K=128): s = Acomb(2048x128) x Bmat(128xN)
//   K rows 0..63  : basInv (cos,sin)/N    -> spectral conv irfft
//   K rows 64..127: current v[b] channels -> bypass 1x1 conv
// The 64x64 v panel is transposed through LDS once per block. Adds bias,
// writes s (f32), accumulates per-row sum/sumsq (instance norm) via atomics.
// Grid: x = N/64 column groups, y = row tiles; 1 wave per block.
// ---------------------------------------------------------------------------
__global__ void fused_spec_bypass_kernel(const _Float16* __restrict__ vH,
                                         const _Float16* __restrict__ Acomb,
                                         const _Float16* __restrict__ basInv,
                                         const float* __restrict__ bb,
                                         float* __restrict__ sF,
                                         float* __restrict__ stats, int layer) {
  __shared__ __align__(16) _Float16 ldsV[64 * 72];   // [n][i] padded (9KB)
  const int cg      = blockIdx.x;          // n0 = cg*64
  const int rowTile = blockIdx.y;
  const int l  = threadIdx.x;
  const int lm = l & 15;
  const int kg = l >> 4;
  const int r0 = rowTile * 16;
  const int b  = r0 >> 6;                  // tile lies in one batch sample
  const int n0 = cg * 64;

  // Stage v panel: 64 channel rows x 64 points (coalesced), transpose in LDS
#pragma unroll
  for (int iRow = l; iRow < W_; iRow += 32) {
    const _Float16* vp = vH + (size_t)(b * W_ + iRow) * N_ + n0;
    v8h c[8];
#pragma unroll
    for (int cix = 0; cix < 8; ++cix)
      c[cix] = *reinterpret_cast<const v8h*>(vp + cix * 8);
#pragma unroll
    for (int n = 0; n < 64; ++n) ldsV[n * 72 + iRow] = c[n >> 3][n & 7];
  }

  // Preload 4 A fragments (K-steps of 32 over K=128); reused across col tiles
  v16h Af[4];
  {
    const _Float16* ar = Acomb + (size_t)(r0 + lm) * 128;
#pragma unroll
    for (int s = 0; s < 4; ++s) {
      v8h a0 = *reinterpret_cast<const v8h*>(ar + s * 32 + kg * 16);
      v8h a1 = *reinterpret_cast<const v8h*>(ar + s * 32 + kg * 16 + 8);
      Af[s] = pack16(a0, a1);
    }
  }
  __syncthreads();   // single-wave WG -> S_NOP

  float ls[8], lq[8];
#pragma unroll
  for (int r = 0; r < 8; ++r) { ls[r] = 0.0f; lq[r] = 0.0f; }

#pragma unroll
  for (int t = 0; t < 4; ++t) {
    const int n0t = n0 + t * 16;
    // Load all four B fragments up front so waits are batched.
    const _Float16* bp0 = basInv + (size_t)(n0t + lm) * 64 + kg * 16;
    v8h i00 = *reinterpret_cast<const v8h*>(bp0);
    v8h i01 = *reinterpret_cast<const v8h*>(bp0 + 8);
    v8h i10 = *reinterpret_cast<const v8h*>(bp0 + 32);
    v8h i11 = *reinterpret_cast<const v8h*>(bp0 + 40);
    const _Float16* vb = ldsV + (t * 16 + lm) * 72 + kg * 16;
    v8h v00 = *reinterpret_cast<const v8h*>(vb);
    v8h v01 = *reinterpret_cast<const v8h*>(vb + 8);
    v8h v10 = *reinterpret_cast<const v8h*>(vb + 32);
    v8h v11 = *reinterpret_cast<const v8h*>(vb + 40);

    v8f acc = (v8f)0.0f;
    acc = wmma_f16(Af[0], pack16(i00, i01), acc);
    acc = wmma_f16(Af[1], pack16(i10, i11), acc);
    acc = wmma_f16(Af[2], pack16(v00, v01), acc);
    acc = wmma_f16(Af[3], pack16(v10, v11), acc);

    // Epilogue: +bias, store s, gather norm statistics
    const int nCol = n0t + lm;
#pragma unroll
    for (int r = 0; r < 8; ++r) {
      const int row = r0 + r + kg * 8;
      const int o = row & 63;
      float val = acc[r] + bb[layer * W_ + o];
      sF[(size_t)row * N_ + nCol] = val;
      ls[r] += val;
      lq[r] += val * val;
    }
  }
  // Reduce within the 16-lane column groups, one atomic per row per block
#pragma unroll
  for (int r = 0; r < 8; ++r) {
#pragma unroll
    for (int off = 1; off < 16; off <<= 1) {
      ls[r] += __shfl_xor(ls[r], off, 32);
      lq[r] += __shfl_xor(lq[r], off, 32);
    }
  }
  if (lm == 0) {
#pragma unroll
    for (int r = 0; r < 8; ++r) {
      const int row = r0 + r + kg * 8;
      atomicAdd(&stats[row], ls[r]);
      atomicAdd(&stats[ROWS_ + row], lq[r]);
    }
  }
}

// ---------------------------------------------------------------------------
// InstanceNorm (affine) + exact GELU; writes next-layer activations in f16.
// One block per (b, channel) row.
// ---------------------------------------------------------------------------
__global__ void norm_gelu_kernel(const float* __restrict__ sF,
                                 const float* __restrict__ stats,
                                 const float* __restrict__ g,
                                 const float* __restrict__ bta,
                                 _Float16* __restrict__ vH, int layer) {
  const int row = blockIdx.x;
  const float mu  = stats[row] * (1.0f / (float)N_);
  const float var = stats[ROWS_ + row] * (1.0f / (float)N_) - mu * mu;
  const float rs  = rsqrtf(var + EPS_);
  const int o = row & 63;
  const float gg = g[layer * W_ + o] * rs;
  const float bb2 = bta[layer * W_ + o];
  const float4* src = reinterpret_cast<const float4*>(sF + (size_t)row * N_);
  _Float16* dst = vH + (size_t)row * N_;
  for (int j = threadIdx.x; j < N_ / 4; j += blockDim.x) {
    float4 x = src[j];
    v4h h;
    h[0] = (_Float16)gelu_exact((x.x - mu) * gg + bb2);
    h[1] = (_Float16)gelu_exact((x.y - mu) * gg + bb2);
    h[2] = (_Float16)gelu_exact((x.z - mu) * gg + bb2);
    h[3] = (_Float16)gelu_exact((x.w - mu) * gg + bb2);
    *reinterpret_cast<v4h*>(dst + j * 4) = h;
  }
}

// ---------------------------------------------------------------------------
// Fused projection: h = gelu(vT(16n x 64w) @ W1 + b1); out = h @ w2 + b2.
// One wave per 16-point tile; 16 WMMAs (8 hidden tiles x 2 K-steps).
// ---------------------------------------------------------------------------
__global__ void proj_kernel(const _Float16* __restrict__ vH,
                            const _Float16* __restrict__ W1T,
                            const float* __restrict__ b1,
                            const float* __restrict__ w2,
                            const float* __restrict__ b2,
                            float* __restrict__ out) {
  __shared__ __align__(16) _Float16 ldsA[16 * 72];   // [n][w] padded
  const int tile = blockIdx.x;            // 0..(B_*N_/16 - 1)
  const int b  = tile >> 9;               // 512 tiles per batch sample
  const int n0 = (tile & 511) * 16;
  const int l  = threadIdx.x;
  const int lm = l & 15;
  const int kg = l >> 4;

  // Stage vT tile: 64 channel rows, 16 points each (coalesced), LDS transpose
#pragma unroll
  for (int wRow = l; wRow < W_; wRow += 32) {
    const _Float16* vp = vH + (size_t)(b * W_ + wRow) * N_ + n0;
    v8h c0 = *reinterpret_cast<const v8h*>(vp);
    v8h c1 = *reinterpret_cast<const v8h*>(vp + 8);
#pragma unroll
    for (int n = 0; n < 8; ++n) ldsA[n * 72 + wRow] = c0[n];
#pragma unroll
    for (int n = 0; n < 8; ++n) ldsA[(8 + n) * 72 + wRow] = c1[n];
  }
  __syncthreads();

  v16h Af[2];
#pragma unroll
  for (int s = 0; s < 2; ++s) {
    v8h a0 = *reinterpret_cast<const v8h*>(ldsA + lm * 72 + s * 32 + kg * 16);
    v8h a1 = *reinterpret_cast<const v8h*>(ldsA + lm * 72 + s * 32 + kg * 16 + 8);
    Af[s] = pack16(a0, a1);
  }

  float accOut[8];
#pragma unroll
  for (int r = 0; r < 8; ++r) accOut[r] = 0.0f;

#pragma unroll
  for (int t = 0; t < 8; ++t) {
    const int col = t * 16 + lm;          // hidden index for this lane
    const _Float16* bp = W1T + (size_t)col * 64 + kg * 16;
    v8h b00 = *reinterpret_cast<const v8h*>(bp);
    v8h b01 = *reinterpret_cast<const v8h*>(bp + 8);
    v8h b10 = *reinterpret_cast<const v8h*>(bp + 32);
    v8h b11 = *reinterpret_cast<const v8h*>(bp + 40);
    v8f acc = (v8f)0.0f;
    acc = wmma_f16(Af[0], pack16(b00, b01), acc);
    acc = wmma_f16(Af[1], pack16(b10, b11), acc);
    const float bias1 = b1[col];
    const float wv = w2[col];
#pragma unroll
    for (int r = 0; r < 8; ++r) {
      accOut[r] += gelu_exact(acc[r] + bias1) * wv;
    }
  }
#pragma unroll
  for (int r = 0; r < 8; ++r) {
#pragma unroll
    for (int off = 1; off < 16; off <<= 1)
      accOut[r] += __shfl_xor(accOut[r], off, 32);
  }
  if (lm == 0) {
    const float bias2 = b2[0];
#pragma unroll
    for (int r = 0; r < 8; ++r) {
      out[(size_t)b * N_ + n0 + r + kg * 8] = accOut[r] + bias2;
    }
  }
}

// ---------------------------------------------------------------------------
extern "C" void kernel_launch(void* const* d_in, const int* in_sizes, int n_in,
                              void* d_out, int out_size, void* d_ws, size_t ws_size,
                              hipStream_t stream) {
  (void)in_sizes; (void)n_in; (void)out_size; (void)ws_size;
  const float* params  = (const float*)d_in[0];
  const float* coords  = (const float*)d_in[1];
  const float* lift_w  = (const float*)d_in[2];
  const float* lift_b  = (const float*)d_in[3];
  const float* spec_wr = (const float*)d_in[4];
  const float* spec_wi = (const float*)d_in[5];
  const float* byp_w   = (const float*)d_in[6];
  const float* byp_b   = (const float*)d_in[7];
  const float* norm_g  = (const float*)d_in[8];
  const float* norm_b  = (const float*)d_in[9];
  const float* proj_w1 = (const float*)d_in[10];
  const float* proj_b1 = (const float*)d_in[11];
  const float* proj_w2 = (const float*)d_in[12];
  const float* proj_b2 = (const float*)d_in[13];
  float* outp = (float*)d_out;

  // Workspace carve (~103 MiB)
  char* ws = (char*)d_ws;
  _Float16* vH = (_Float16*)ws;          ws += (size_t)ROWS_ * N_ * 2;          // 32 MiB
  float*    sF = (float*)ws;             ws += (size_t)ROWS_ * N_ * 4;          // 64 MiB
  float*    ft = (float*)ws;             ws += (size_t)ROWS_ * 64 * 4;          // 512 KiB
  float*    ftPart = (float*)ws;         ws += (size_t)KPART_ * ROWS_ * 64 * 4; // 4 MiB
  _Float16* Acomb = (_Float16*)ws;       ws += (size_t)ROWS_ * 128 * 2;         // 512 KiB
  _Float16* basT  = (_Float16*)ws;       ws += (size_t)64 * N_ * 2;             // 1 MiB
  _Float16* basInv = (_Float16*)ws;      ws += (size_t)N_ * 64 * 2;             // 1 MiB
  _Float16* W1T = (_Float16*)ws;         ws += (size_t)128 * 64 * 2;            // 16 KiB
  float*    stats = (float*)ws;          ws += (size_t)ROWS_ * 2 * 4;           // 16 KiB

  init_basis_kernel<<<(N_ * 64) / 256, 256, 0, stream>>>(basT, basInv);
  prep_w1t_kernel<<<(128 * 64) / 256, 256, 0, stream>>>(proj_w1, W1T);
  lift_kernel<<<(B_ * N_) / 256, 256, 0, stream>>>(params, coords, lift_w, lift_b, vH);

  for (int l = 0; l < L_; ++l) {
    dft_fwd_kernel<<<dim3(ROWS_ / 16, KPART_), 32, 0, stream>>>(vH, basT, ftPart);
    reduce_ft_kernel<<<(ROWS_ * 64) / 256, 256, 0, stream>>>(ftPart, ft);
    mix_kernel<<<(B_ * W_ * M_) / 256, 256, 0, stream>>>(ft, spec_wr, spec_wi,
                                                         byp_w, Acomb, stats, l);
    fused_spec_bypass_kernel<<<dim3(N_ / 64, ROWS_ / 16), 32, 0, stream>>>(
        vH, Acomb, basInv, byp_b, sF, stats, l);
    norm_gelu_kernel<<<ROWS_, 256, 0, stream>>>(sF, stats, norm_g, norm_b, vH, l);
  }

  proj_kernel<<<(B_ * N_) / 16, 32, 0, stream>>>(vH, W1T, proj_b1, proj_w2,
                                                 proj_b2, outp);
}
